// Attn_30107720745210
// MI455X (gfx1250) — matrix operand order repaired and checked by reading, not compiled, not run on hardware
//
#include <hip/hip_runtime.h>
#include <hip/hip_bf16.h>

// ---------------------------------------------------------------------------
// Problem constants (match reference)
// ---------------------------------------------------------------------------
#define BBATCH 64
#define LL1    512
#define LL2    512
#define DOUT   1024
#define DATTN  512

typedef __attribute__((ext_vector_type(16))) __bf16 v16bf;
typedef __attribute__((ext_vector_type(8)))  float  v8f;

__device__ __forceinline__ __bf16 to_bf16(float f) {
  unsigned u = __float_as_uint(f);
  unsigned r = (u + 0x7FFFu + ((u >> 16) & 1u)) >> 16;   // RNE
  unsigned short s = (unsigned short)r;
  __bf16 h; __builtin_memcpy(&h, &s, sizeof(h)); return h;
}

__device__ __forceinline__ unsigned short bf16_bits(float f) {
  unsigned u = __float_as_uint(f);
  return (unsigned short)((u + 0x7FFFu + ((u >> 16) & 1u)) >> 16);
}

// ---------------------------------------------------------------------------
// f32 -> bf16 bulk convert
// ---------------------------------------------------------------------------
__global__ void f32_to_bf16_kernel(const float* __restrict__ in,
                                   unsigned short* __restrict__ out,
                                   long long n) {
  long long i = (long long)blockIdx.x * blockDim.x + threadIdx.x;
  long long stride = (long long)gridDim.x * blockDim.x;
  for (; i < n; i += stride) out[i] = bf16_bits(in[i]);
}

// f32 [R][C] -> bf16 transposed [C][R]  (small weight matrices only)
__global__ void f32_to_bf16_T_kernel(const float* __restrict__ in,
                                     unsigned short* __restrict__ out,
                                     int R, int C) {
  long long n = (long long)R * C;
  long long i = (long long)blockIdx.x * blockDim.x + threadIdx.x;
  long long stride = (long long)gridDim.x * blockDim.x;
  for (; i < n; i += stride) {
    int r = (int)(i / C), c = (int)(i % C);
    out[(size_t)c * R + r] = bf16_bits(in[i]);
  }
}

// ---------------------------------------------------------------------------
// bf16 WMMA GEMM  (block tile 128x64, K-step 32, 8 wave32 waves)
//   A: row-major [M][K]   (k-contiguous)
//   B: stored  [N][K]     (k-contiguous; element (k,n) = B[n*ldb + k])
// Register-staged double-buffered LDS pipeline: one barrier per K-step,
// global loads of tile kt+1 overlap WMMA of tile kt.
//
// EPI: 0 = store bf16 (row-major)
//      1 = tanh -> bf16 row-major (outB) + bf16 transposed (outBT)
//      2 = f32 row-major (outF)  + bf16 transposed (outBT)
//      3 = fused score: h = tanh(init + A@B); partial[row, xblk] = sum_n h*wvec[n]
// ---------------------------------------------------------------------------
#define BM 128
#define BN 64
#define BK 32
#define LDK_S 40   // k-stride in LDS: 80B, multiple of 16 -> aligned b128 ops
#define ASZ (BM * LDK_S)
#define BSZ (BN * LDK_S)

template <int EPI>
__global__ __launch_bounds__(256, 2)
void gemm_bf16_wmma(const __bf16* __restrict__ A, long long strideA, int lda,
                    const __bf16* __restrict__ Bm, long long strideB, int ldb,
                    int M, int N, int K,
                    __bf16* __restrict__ outB, long long strideOutB,
                    __bf16* __restrict__ outBT, long long strideOutBT, int ltShift,
                    float* __restrict__ outF, long long strideOutF,
                    const float* __restrict__ initF, long long strideInit,
                    const float* __restrict__ wvec,
                    float* __restrict__ partial, int P)
{
  __shared__ __align__(16) __bf16 As[2 * ASZ];
  __shared__ __align__(16) __bf16 Bs[2 * BSZ];
  __shared__ float sred[BM * 2];

  const int tid  = threadIdx.x;
  const int lane = tid & 31;
  const int wave = tid >> 5;
  const int bz   = blockIdx.z;
  const int row0 = blockIdx.y * BM;
  const int col0 = blockIdx.x * BN;

  const __bf16* Ab = A  + (size_t)bz * strideA;
  const __bf16* Bb = Bm + (size_t)bz * strideB;

  const int wm0   = (wave >> 1) * 32;          // wave tile row in block tile
  const int wn0   = (wave & 1) * 32;           // wave tile col in block tile
  const int lrow  = lane & 15;
  const int ahalf = (lane < 16) ? 0 : 8;       // A frag k sub-base (ISA layout)
  const int bbase = (lane < 16) ? 0 : 16;      // unused; B frag loads use ahalf like A
  const int chalf = (lane >> 4) << 3;          // C/D frag row offset (0 or 8)

  // staging coordinates (A: 32 contiguous bytes per thread; B: 16 bytes)
  const int sa_r  = tid >> 1;                  // A m-row 0..127
  const int sa_kc = (tid & 1) * 16;            // A k chunk 0 or 16
  const int sb_n  = tid & 63;                  // B n row
  const int sb_kc = (tid >> 6) * 8;            // B k chunk 0,8,16,24

  const __bf16* gA = Ab + (size_t)(row0 + sa_r) * lda + sa_kc;
  const __bf16* gB = Bb + (size_t)(col0 + sb_n) * ldb + sb_kc;

  const v8f zv = {0.f, 0.f, 0.f, 0.f, 0.f, 0.f, 0.f, 0.f};
  v8f acc[2][2] = {{zv, zv}, {zv, zv}};

  const int ksteps = K / BK;

  // prologue: load tile 0 into registers
  uint4 ra0 = *(const uint4*)(gA);
  uint4 ra1 = *(const uint4*)(gA + 8);
  uint4 rb0 = *(const uint4*)(gB);

  for (int kt = 0; kt < ksteps; ++kt) {
    const int buf = kt & 1;
    __bf16* as = As + buf * ASZ;
    __bf16* bs = Bs + buf * BSZ;

    // ---- store staged registers into LDS[buf] ----
    *(uint4*)&as[sa_r * LDK_S + sa_kc]     = ra0;
    *(uint4*)&as[sa_r * LDK_S + sa_kc + 8] = ra1;
    *(uint4*)&bs[sb_n * LDK_S + sb_kc]     = rb0;
    __syncthreads();

    // ---- issue global loads for tile kt+1 (overlap with compute) ----
    if (kt + 1 < ksteps) {
      const int k1 = (kt + 1) * BK;
      ra0 = *(const uint4*)(gA + k1);
      ra1 = *(const uint4*)(gA + k1 + 8);
      rb0 = *(const uint4*)(gB + k1);
      if (kt + 2 < ksteps) {
        __builtin_prefetch((const void*)(gA + k1 + BK), 0, 1);
        __builtin_prefetch((const void*)(gB + k1 + BK), 0, 1);
      }
    }

    // ---- fragment loads: 2x aligned ds_load_b128 per fragment ----
    v16bf afrag[2], bfrag[2];
#pragma unroll
    for (int wm = 0; wm < 2; ++wm) {
      const __bf16* ap = &as[(wm0 + wm * 16 + lrow) * LDK_S];
      uint4 lo = *(const uint4*)(ap + ahalf);
      uint4 hi = *(const uint4*)(ap + 16 + ahalf);
      __builtin_memcpy(&afrag[wm], &lo, 16);
      __builtin_memcpy((char*)&afrag[wm] + 16, &hi, 16);
    }
#pragma unroll
    for (int wn = 0; wn < 2; ++wn) {
      const __bf16* bp = &bs[(wn0 + wn * 16 + lrow) * LDK_S];
      uint4 lo = *(const uint4*)(bp + ahalf);
      uint4 hi = *(const uint4*)(bp + 16 + ahalf);
      __builtin_memcpy(&bfrag[wn], &lo, 16);
      __builtin_memcpy((char*)&bfrag[wn] + 16, &hi, 16);
    }

    // ---- 4x v_wmma_f32_16x16x32_bf16 per K-step per wave ----
#pragma unroll
    for (int wm = 0; wm < 2; ++wm)
#pragma unroll
      for (int wn = 0; wn < 2; ++wn)
        acc[wm][wn] = __builtin_amdgcn_wmma_f32_16x16x32_bf16(
            false, afrag[wm], false, bfrag[wn],
            (short)0, acc[wm][wn], false, false);
  }

  // ---- epilogue ----
  if (EPI <= 2) {
    const int ltMask = (1 << ltShift) - 1;
#pragma unroll
    for (int wm = 0; wm < 2; ++wm) {
#pragma unroll
      for (int wn = 0; wn < 2; ++wn) {
        int rbase = row0 + wm0 + wm * 16 + chalf;
        int col   = col0 + wn0 + wn * 16 + lrow;
#pragma unroll
        for (int i = 0; i < 8; ++i) {
          int row = rbase + i;
          float v = acc[wm][wn][i];
          if (EPI == 1) v = tanhf(v);
          if (EPI == 0 || EPI == 1) {
            outB[(size_t)bz * strideOutB + (size_t)row * N + col] = to_bf16(v);
          }
          if (EPI == 2) {
            outF[(size_t)bz * strideOutF + (size_t)row * N + col] = v;
          }
          if (EPI == 1 || EPI == 2) {
            size_t ti = (((size_t)(row >> ltShift) * N + col) << ltShift) + (row & ltMask);
            outBT[(size_t)bz * strideOutBT + ti] = to_bf16(v);
          }
        }
      }
    }
  } else {
    // fused: h = tanh(init + acc); row-partial = sum_cols h * wvec[col]
#pragma unroll
    for (int wm = 0; wm < 2; ++wm) {
      int rbase = row0 + wm0 + wm * 16 + chalf;
#pragma unroll
      for (int i = 0; i < 8; ++i) {
        int row = rbase + i;
        float sum = 0.f;
#pragma unroll
        for (int wn = 0; wn < 2; ++wn) {
          int col = col0 + wn0 + wn * 16 + lrow;
          float ini = initF[(size_t)bz * strideInit + (size_t)row * N + col];
          float h = tanhf(acc[wm][wn][i] + ini);
          sum += h * wvec[col];
        }
#pragma unroll
        for (int off = 1; off < 16; off <<= 1)
          sum += __shfl_xor(sum, off, 32);
        if (lrow == 0) {
          int rl = wm0 + wm * 16 + chalf + i;     // local row 0..127
          sred[rl * 2 + (wave & 1)] = sum;
        }
      }
    }
    __syncthreads();
    if (tid < BM) {
      float t = sred[tid * 2] + sred[tid * 2 + 1];
      partial[((size_t)bz * M + row0 + tid) * P + blockIdx.x] = t;
    }
  }
}

// ---------------------------------------------------------------------------
// masked softmax, faithful to reference: r = softmax(s*m); r *= m; r /= (sum(r)+1e-13)
// blockDim.x == L == 512, one block per batch row
// ---------------------------------------------------------------------------
__global__ void masked_softmax_kernel(const float* __restrict__ partial, int P,
                                      const int* __restrict__ mask,
                                      float* __restrict__ out, int L) {
  int b = blockIdx.x, t = threadIdx.x;
  __shared__ float red[512];
  float s = 0.f;
  for (int p = 0; p < P; ++p) s += partial[((size_t)b * L + t) * P + p];
  float m = (float)mask[(size_t)b * L + t];
  float x = s * m;

  red[t] = x; __syncthreads();
  for (int o = 256; o > 0; o >>= 1) { if (t < o) red[t] = fmaxf(red[t], red[t + o]); __syncthreads(); }
  float mx = red[0]; __syncthreads();

  float e = __expf(x - mx);
  red[t] = e; __syncthreads();
  for (int o = 256; o > 0; o >>= 1) { if (t < o) red[t] += red[t + o]; __syncthreads(); }
  float denom = red[0]; __syncthreads();

  float r = (e / denom) * m;
  red[t] = r; __syncthreads();
  for (int o = 256; o > 0; o >>= 1) { if (t < o) red[t] += red[t + o]; __syncthreads(); }
  float rs = red[0];

  out[(size_t)b * L + t] = r / (rs + 1e-13f);
}

// ---------------------------------------------------------------------------
// out[b,d] = sum_l attn[b,l] * x[b,l,d]   (f32, bandwidth bound)
// ---------------------------------------------------------------------------
__global__ void weighted_sum_kernel(const float* __restrict__ attn,
                                    const float* __restrict__ x,
                                    float* __restrict__ out, int L, int D) {
  int b = blockIdx.y;
  int d = blockIdx.x * blockDim.x + threadIdx.x;
  __shared__ float aw[512];
  for (int i = threadIdx.x; i < L; i += blockDim.x) aw[i] = attn[(size_t)b * L + i];
  __syncthreads();
  const float* xb = x + (size_t)b * L * D + d;
  float acc = 0.f;
  for (int l = 0; l < L; ++l) acc += aw[l] * xb[(size_t)l * D];
  out[(size_t)b * D + d] = acc;
}

// ---------------------------------------------------------------------------
// Launch sequence
// ---------------------------------------------------------------------------
extern "C" void kernel_launch(void* const* d_in, const int* in_sizes, int n_in,
                              void* d_out, int out_size, void* d_ws, size_t ws_size,
                              hipStream_t stream) {
  (void)in_sizes; (void)n_in; (void)out_size; (void)ws_size;
  const float* x1    = (const float*)d_in[0];
  const float* x2    = (const float*)d_in[1];
  const int*   mask1 = (const int*)d_in[2];
  const int*   mask2 = (const int*)d_in[3];
  const float* W     = (const float*)d_in[4];
  const float* Wv    = (const float*)d_in[5];
  const float* Wq    = (const float*)d_in[6];
  const float* w_hv  = (const float*)d_in[7];
  const float* w_hq  = (const float*)d_in[8];
  float* out = (float*)d_out;

  char* ws = (char*)d_ws;
  size_t off = 0;
  auto alloc = [&](size_t bytes) -> char* {
    char* p = ws + off; off += (bytes + 255) & ~(size_t)255; return p;
  };

  const long long nX = (long long)BBATCH * LL1 * DOUT;   // x1/x2 element count
  __bf16* x1b     = (__bf16*)alloc((size_t)nX * 2);
  __bf16* x2b     = (__bf16*)alloc((size_t)nX * 2);
  __bf16* Wt      = (__bf16*)alloc((size_t)DOUT * DOUT * 2);            // W^T  [N][K]
  __bf16* Wvt     = (__bf16*)alloc((size_t)DOUT * DATTN * 2);           // Wv^T [A][OUT]
  __bf16* Wqt     = (__bf16*)alloc((size_t)DOUT * DATTN * 2);           // Wq^T
  __bf16* T1b     = (__bf16*)alloc((size_t)nX * 2);                     // x1 @ W (bf16)
  __bf16* Cb      = (__bf16*)alloc((size_t)BBATCH * LL1 * LL2 * 2);     // tanh affinity
  __bf16* CbT     = (__bf16*)alloc((size_t)BBATCH * LL1 * LL2 * 2);     // C^T per batch
  float*  x1Wv_f  = (float*)alloc((size_t)BBATCH * LL1 * DATTN * 4);
  float*  x2Wq_f  = (float*)alloc((size_t)BBATCH * LL2 * DATTN * 4);
  __bf16* x1Wv_bT = (__bf16*)alloc((size_t)BBATCH * LL1 * DATTN * 2);   // [B][A][L1]
  __bf16* x2Wq_bT = (__bf16*)alloc((size_t)BBATCH * LL2 * DATTN * 2);   // [B][A][L2]
  float*  pv      = (float*)alloc((size_t)BBATCH * LL1 * 8 * 4);        // score partials
  float*  pq      = (float*)alloc((size_t)BBATCH * LL2 * 8 * 4);
  float*  av      = (float*)alloc((size_t)BBATCH * LL1 * 4);
  float*  aq      = (float*)alloc((size_t)BBATCH * LL2 * 4);

  // 1) converts (weights transposed so every GEMM's B operand is k-contiguous)
  f32_to_bf16_kernel<<<4096, 256, 0, stream>>>(x1, (unsigned short*)x1b, nX);
  f32_to_bf16_kernel<<<4096, 256, 0, stream>>>(x2, (unsigned short*)x2b, nX);
  f32_to_bf16_T_kernel<<<1024, 256, 0, stream>>>(W,  (unsigned short*)Wt,  DOUT, DOUT);
  f32_to_bf16_T_kernel<<<512,  256, 0, stream>>>(Wv, (unsigned short*)Wvt, DOUT, DATTN);
  f32_to_bf16_T_kernel<<<512,  256, 0, stream>>>(Wq, (unsigned short*)Wqt, DOUT, DATTN);

  dim3 blk(256);

  // 2) T1 = x1 @ W   (merged M: 32768 x 1024 x 1024, B = W^T)
  {
    int M = BBATCH * LL1, N = DOUT, K = DOUT;
    dim3 grid(N / BN, M / BM, 1);
    gemm_bf16_wmma<0><<<grid, blk, 0, stream>>>(x1b, 0, DOUT, Wt, 0, DOUT,
        M, N, K, T1b, 0, nullptr, 0, 9, nullptr, 0, nullptr, 0, nullptr, nullptr, 0);
  }
  // 3) x1Wv = x1 @ Wv : f32 row-major + bf16 transposed per batch [A][L1]
  {
    int M = BBATCH * LL1, N = DATTN, K = DOUT;
    dim3 grid(N / BN, M / BM, 1);
    gemm_bf16_wmma<2><<<grid, blk, 0, stream>>>(x1b, 0, DOUT, Wvt, 0, DOUT,
        M, N, K, nullptr, 0, x1Wv_bT, 0, /*ltShift=*/9, x1Wv_f, 0,
        nullptr, 0, nullptr, nullptr, 0);
  }
  // 4) x2Wq = x2 @ Wq : dual
  {
    int M = BBATCH * LL2, N = DATTN, K = DOUT;
    dim3 grid(N / BN, M / BM, 1);
    gemm_bf16_wmma<2><<<grid, blk, 0, stream>>>(x2b, 0, DOUT, Wqt, 0, DOUT,
        M, N, K, nullptr, 0, x2Wq_bT, 0, /*ltShift=*/9, x2Wq_f, 0,
        nullptr, 0, nullptr, nullptr, 0);
  }
  // 5) C = tanh(T1 @ x2^T)  batched; B = x2 (already [L2][OUT], k-contiguous)
  //    stores C row-major (A of step 6) and C^T (A of step 7)
  {
    int M = LL1, N = LL2, K = DOUT;
    dim3 grid(N / BN, M / BM, BBATCH);
    gemm_bf16_wmma<1><<<grid, blk, 0, stream>>>(
        T1b, (long long)LL1 * DOUT, DOUT,
        x2b, (long long)LL2 * DOUT, DOUT,
        M, N, K, Cb, (long long)LL1 * LL2,
        CbT, (long long)LL1 * LL2, /*ltShift=*/9,
        nullptr, 0, nullptr, 0, nullptr, nullptr, 0);
  }
  // 6) fused: Hv = tanh(x1Wv + C @ x2Wq); pv = row-partials of Hv @ w_hv
  {
    int M = LL1, N = DATTN, K = LL2;
    dim3 grid(N / BN, M / BM, BBATCH);
    gemm_bf16_wmma<3><<<grid, blk, 0, stream>>>(
        Cb, (long long)LL1 * LL2, LL2,
        x2Wq_bT, (long long)LL2 * DATTN, LL2,
        M, N, K, nullptr, 0, nullptr, 0, 9, nullptr, 0,
        x1Wv_f, (long long)LL1 * DATTN, w_hv, pv, N / BN);
  }
  // 7) fused: Hq = tanh(x2Wq + C^T @ x1Wv); pq = row-partials of Hq @ w_hq
  //    A = C^T (row-major [L2][L1]), B = x1Wv^T ([A][L1])
  {
    int M = LL2, N = DATTN, K = LL1;
    dim3 grid(N / BN, M / BM, BBATCH);
    gemm_bf16_wmma<3><<<grid, blk, 0, stream>>>(
        CbT, (long long)LL1 * LL2, LL1,
        x1Wv_bT, (long long)LL1 * DATTN, LL1,
        M, N, K, nullptr, 0, nullptr, 0, 9, nullptr, 0,
        x2Wq_f, (long long)LL2 * DATTN, w_hq, pq, N / BN);
  }

  // 8) masked softmax
  masked_softmax_kernel<<<BBATCH, 512, 0, stream>>>(pv, 8, mask1, av, LL1);
  masked_softmax_kernel<<<BBATCH, 512, 0, stream>>>(pq, 8, mask2, aq, LL2);

  // 9) v_hat / q_hat against original f32 features
  {
    dim3 grid(DOUT / 256, BBATCH);
    weighted_sum_kernel<<<grid, 256, 0, stream>>>(av, x1, out, LL1, DOUT);
    weighted_sum_kernel<<<grid, 256, 0, stream>>>(aq, x2, out + (size_t)BBATCH * DOUT, LL2, DOUT);
  }
}
